// BlindDeblurring_56573309224033
// MI455X (gfx1250) — matrix operand order repaired and checked by reading, not compiled, not run on hardware
//
#include <hip/hip_runtime.h>
#include <math.h>

#define NPIX   3071
#define NPIX2  (NPIX * NPIX)          // 9431041
#define KS     11
#define MFFT   8192
#define LOG2M  13
#define EPSF   1e-10f
#define CH     1530                   // (NPIX - KS) / 2
#define PI_F   3.14159265358979323846f
#define NUM_ITERS 20
#define RL_ITERS  5

// bf16 image buffer uses padded pitch so rows are 16B aligned
#define XBF_W  3072

typedef __attribute__((ext_vector_type(16))) __bf16 v16bf;
typedef __attribute__((ext_vector_type(8)))  float  v8f;

union Frag16 { unsigned int u[8]; v16bf v; };

__device__ __forceinline__ unsigned short f2bf(float f) {
  unsigned int u = __float_as_uint(f);
  unsigned int r = u + 0x7FFFu + ((u >> 16) & 1u);   // round-to-nearest-even
  return (unsigned short)(r >> 16);
}

__device__ __forceinline__ float2 cmul(float2 a, float2 b) {
  return make_float2(a.x * b.x - a.y * b.y, a.x * b.y + a.y * b.x);
}

// ---------------------------------------------------------------------------
// 11x11 "same" correlation via WMMA implicit-GEMM (bf16 in, f32 out).
// One 16x16 output tile per wave; 8 waves (8 tiles in x) per 256-thread block.
// Shifted Toeplitz: A[m,k] = w[dy][k-m-1]  (k in [m+1, m+11])
//                   B[k,n] = in[y0+n+dy-5][x0+k-6]
// => D[m,n] = sum_{dy,dx} w[dy][dx] * in[y0+n+dy-5][x0+m+dx-5]
// Left halo of 6 keeps every global dword 4B-aligned (async-to-LDS legal) and
// every LDS B-fragment read 16B-aligned (ds_load_b128).
// ---------------------------------------------------------------------------
#define TILES_PB    8
#define PADL        6     // left/top-left halo (x); y halo stays 5
#define PATCH_ROWS  26
#define PATCH_COLS  144   // gx in [colBase-6, colBase+137]
#define PATCH_W     152   // pitch (shorts): 304B rows, 16B aligned, conflict-free stride

__global__ __launch_bounds__(256)
void k_conv_wmma(const unsigned short* __restrict__ src, float* __restrict__ dst,
                 const float* __restrict__ wker) {
  __shared__ __align__(16) unsigned short patch[PATCH_ROWS * PATCH_W];
  __shared__ __align__(16) unsigned int   afrag[KS * 32 * 8]; // [dy][lane][v] packed bf16x2
  __shared__ unsigned short wbf[KS * KS];

  const int tid     = threadIdx.x;
  const int colBase = blockIdx.x * (TILES_PB * 16);
  const int rowBase = blockIdx.y * 16;

  if (tid < KS * KS) wbf[tid] = f2bf(wker[tid]);

  // Interior blocks: all staged addresses in-bounds -> async global->LDS copy.
  const bool interior = (blockIdx.x >= 1) & (blockIdx.x <= 22) &
                        (blockIdx.y >= 1) & (blockIdx.y <= 190);
  if (interior) {
    // 26 rows x 72 dwords (144 bf16) per block
    for (int i = tid; i < PATCH_ROWS * (PATCH_COLS / 2); i += 256) {
      int r  = i / (PATCH_COLS / 2);
      int c  = i % (PATCH_COLS / 2);            // dword index within row
      int gy = rowBase + r - 5;
      int gx = colBase + 2 * c - PADL;          // even -> 4B aligned
      unsigned long long ga =
          (unsigned long long)(const void*)(src + (size_t)gy * XBF_W + gx);
      unsigned int lo =
          (unsigned int)(uintptr_t)(const void*)&patch[r * PATCH_W + 2 * c];
      asm volatile("global_load_async_to_lds_b32 %0, %1, off"
                   :: "v"(lo), "v"(ga) : "memory");
    }
    asm volatile("s_wait_asynccnt 0x0" ::: "memory");
  } else {
    // Border blocks: masked scalar staging with zero padding
    for (int i = tid; i < PATCH_ROWS * PATCH_COLS; i += 256) {
      int r = i / PATCH_COLS, c = i % PATCH_COLS;
      int gy = rowBase + r - 5;
      int gx = colBase + c - PADL;
      unsigned short v = 0;
      if (gy >= 0 && gy < NPIX && gx >= 0 && gx < NPIX) {
        v = src[(size_t)gy * XBF_W + gx];
        __builtin_prefetch(&src[(size_t)min(gy + 16, NPIX - 1) * XBF_W + gx], 0, 0);
      }
      patch[r * PATCH_W + c] = v;
    }
  }
  __syncthreads();

  // Build shifted-Toeplitz A-fragments for all 11 dy (documented 16-bit A layout:
  // lane -> M=lane%16; V0..3 hold K=2v,2v+1 (+8 for lanes 16..31), V4..7 hold K=16+2(v-4).. (+8))
  for (int i = tid; i < KS * 32 * 8; i += 256) {
    int v    = i & 7;
    int lane = (i >> 3) & 31;
    int dy   = i >> 8;
    int m    = lane & 15;
    int kb   = ((v < 4) ? (2 * v) : (16 + 2 * (v - 4))) + ((lane >= 16) ? 8 : 0);
    unsigned short lo = 0, hi = 0;
    int dxlo = kb - 1 - m;        // dx for k = kb
    int dxhi = kb - m;            // dx for k = kb+1
    if (dxlo >= 0 && dxlo < KS) lo = wbf[dy * KS + dxlo];
    if (dxhi >= 0 && dxhi < KS) hi = wbf[dy * KS + dxhi];
    afrag[i] = (unsigned int)lo | ((unsigned int)hi << 16);
  }
  __syncthreads();

  const int wave = tid >> 5;
  const int lane = tid & 31;
  const int n    = lane & 15;   // N index = output row offset
  const int g    = lane >> 4;   // lane group
  const int x0   = colBase + wave * 16;

  v8f c = {0.f, 0.f, 0.f, 0.f, 0.f, 0.f, 0.f, 0.f};
  for (int dy = 0; dy < KS; ++dy) {
    Frag16 a, b;
    {
      const uint4* ap = (const uint4*)&afrag[(dy * 32 + lane) * 8]; // 32B aligned
      uint4 t0 = ap[0], t1 = ap[1];
      a.u[0] = t0.x; a.u[1] = t0.y; a.u[2] = t0.z; a.u[3] = t0.w;
      a.u[4] = t1.x; a.u[5] = t1.y; a.u[6] = t1.z; a.u[7] = t1.w;
    }
    {
      // B: row (n+dy), patch cols 16*wave + 8*g + [0..7] and 16*wave + 16 + 8*g + [0..7]  (16B aligned)
      const uint4* bp =
          (const uint4*)&patch[(n + dy) * PATCH_W + 16 * wave + 8 * g];
      uint4 t0 = bp[0], t1 = bp[2];
      b.u[0] = t0.x; b.u[1] = t0.y; b.u[2] = t0.z; b.u[3] = t0.w;
      b.u[4] = t1.x; b.u[5] = t1.y; b.u[6] = t1.z; b.u[7] = t1.w;
    }
    c = __builtin_amdgcn_wmma_f32_16x16x32_bf16(false, a.v, false, b.v,
                                                (short)0, c, false, false);
  }

  // D layout: VGPR r, lanes 0-15 -> (M=r, N=lane); lanes 16-31 -> (M=r+8, N=lane-16)
  const int y  = rowBase + n;
  const int xs = x0 + g * 8;
  if (y < NPIX) {
    float* drow = dst + (size_t)y * NPIX + xs;
    if (xs + 7 < NPIX) {
#pragma unroll
      for (int r = 0; r < 8; ++r) drow[r] = c[r];
    } else {
#pragma unroll
      for (int r = 0; r < 8; ++r) if (xs + r < NPIX) drow[r] = c[r];
    }
  }
}

// ---------------------------------------------------------------------------
// Pointwise kernels (xbf uses padded pitch XBF_W)
// ---------------------------------------------------------------------------
__global__ void k_init_ker(float* ker, float* kerflip) {
  int i = threadIdx.x;
  if (i < KS * KS) {
    float v = (i == (KS / 2) * KS + KS / 2) ? 1.f : 0.f;
    ker[i] = v; kerflip[i] = v;
  }
}

__global__ void k_init_from_blurred(const float* __restrict__ blurred,
                                    float* __restrict__ x32,
                                    unsigned short* __restrict__ xbf) {
  int stride = gridDim.x * blockDim.x;
  for (int i = blockIdx.x * blockDim.x + threadIdx.x; i < NPIX2; i += stride) {
    float v = blurred[i];
    x32[i] = v;
    int row = i / NPIX, col = i - row * NPIX;
    xbf[(size_t)row * XBF_W + col] = f2bf(v);
  }
}

__global__ void k_ratio(const float* __restrict__ blurred, const float* __restrict__ est,
                        unsigned short* __restrict__ xbf) {
  int stride = gridDim.x * blockDim.x;
  for (int i = blockIdx.x * blockDim.x + threadIdx.x; i < NPIX2; i += stride) {
    float v = blurred[i] / (est[i] + EPSF);
    int row = i / NPIX, col = i - row * NPIX;
    xbf[(size_t)row * XBF_W + col] = f2bf(v);
  }
}

__global__ void k_update_x(float* __restrict__ x32, const float* __restrict__ corr,
                           unsigned short* __restrict__ xbf) {
  int stride = gridDim.x * blockDim.x;
  for (int i = blockIdx.x * blockDim.x + threadIdx.x; i < NPIX2; i += stride) {
    float v = x32[i] * corr[i];
    v = fminf(fmaxf(v, 0.f), 1.f);
    x32[i] = v;
    int row = i / NPIX, col = i - row * NPIX;
    xbf[(size_t)row * XBF_W + col] = f2bf(v);
  }
}

// ---------------------------------------------------------------------------
// Bluestein FFT machinery (length 3071 via length-8192 circular convolution)
// ---------------------------------------------------------------------------
__global__ void k_chirp_setup(float2* __restrict__ chirpW, float2* __restrict__ bbuf) {
  int i = blockIdx.x * blockDim.x + threadIdx.x;
  if (i >= MFFT) return;
  int m = -1;
  if (i < NPIX) m = i;
  else if (MFFT - i < NPIX) m = MFFT - i;
  float2 bv = make_float2(0.f, 0.f);
  if (m >= 0) {
    long long m2 = ((long long)m * m) % (2LL * NPIX);
    float ang = PI_F * (float)m2 / (float)NPIX;
    float s, c; __sincosf(ang, &s, &c);
    bv = make_float2(c, s);                       // b[m] = e^{+i pi m^2 / N}
  }
  bbuf[i] = bv;
  if (i < NPIX) {
    long long m2 = ((long long)i * i) % (2LL * NPIX);
    float ang = PI_F * (float)m2 / (float)NPIX;
    float s, c; __sincosf(ang, &s, &c);
    chirpW[i] = make_float2(c, -s);               // w[n] = e^{-i pi n^2 / N}
  }
}

// In-place radix-2 DIT FFT of 8192 points in LDS. sign = -1 fwd, +1 inv (unnormalized).
__device__ void fft_lds(float2* buf, int sign) {
  const int tid = threadIdx.x;   // 256 threads
  for (int i = tid; i < MFFT; i += 256) {
    int r = (int)(__brev((unsigned)i) >> (32 - LOG2M));
    if (r > i) { float2 t = buf[i]; buf[i] = buf[r]; buf[r] = t; }
  }
  __syncthreads();
  for (int len = 2; len <= MFFT; len <<= 1) {
    int half = len >> 1;
    float base = (float)sign * 2.f * PI_F / (float)len;
    for (int k = tid; k < MFFT / 2; k += 256) {
      int j  = k & (half - 1);
      int i0 = (k << 1) - j;
      int i1 = i0 + half;
      float s, c; __sincosf(base * (float)j, &s, &c);
      float2 a = buf[i0], b = buf[i1];
      float br = b.x * c - b.y * s;
      float bi = b.x * s + b.y * c;
      buf[i0] = make_float2(a.x + br, a.y + bi);
      buf[i1] = make_float2(a.x - br, a.y - bi);
    }
    __syncthreads();
  }
}

__global__ __launch_bounds__(256)
void k_fft8192(const float2* __restrict__ src, float2* __restrict__ dst, int sign) {
  __shared__ float2 buf[MFFT];
  for (int i = threadIdx.x; i < MFFT; i += 256) buf[i] = src[i];
  __syncthreads();
  fft_lds(buf, sign);
  for (int i = threadIdx.x; i < MFFT; i += 256) dst[i] = buf[i];
}

// One block per row: length-3071 forward DFT of the row via Bluestein.
__global__ __launch_bounds__(256)
void k_bluestein_rows(const void* __restrict__ srcv, int srcIsReal,
                      float2* __restrict__ dst,
                      const float2* __restrict__ chirpW,
                      const float2* __restrict__ Bhat) {
  __shared__ float2 buf[MFFT];   // 64 KB
  const int row = blockIdx.x;
  const int tid = threadIdx.x;
  if (srcIsReal) {
    const float* s = (const float*)srcv + (size_t)row * NPIX;
    for (int i = tid; i < MFFT; i += 256) {
      float2 v = make_float2(0.f, 0.f);
      if (i < NPIX) { float2 w = chirpW[i]; float xr = s[i]; v = make_float2(xr * w.x, xr * w.y); }
      buf[i] = v;
    }
  } else {
    const float2* s = (const float2*)srcv + (size_t)row * NPIX;
    for (int i = tid; i < MFFT; i += 256) {
      float2 v = make_float2(0.f, 0.f);
      if (i < NPIX) v = cmul(s[i], chirpW[i]);
      buf[i] = v;
    }
  }
  __syncthreads();
  fft_lds(buf, -1);
  for (int i = tid; i < MFFT; i += 256) buf[i] = cmul(buf[i], Bhat[i]);
  __syncthreads();
  fft_lds(buf, +1);
  const float scale = 1.f / (float)MFFT;
  float2* d = dst + (size_t)row * NPIX;
  for (int i = tid; i < NPIX; i += 256) {
    float2 v = cmul(buf[i], chirpW[i]);
    d[i] = make_float2(v.x * scale, v.y * scale);
  }
}

__global__ void k_transpose(const float2* __restrict__ src, float2* __restrict__ dst) {
  __shared__ float2 tile[32][33];
  int x = blockIdx.x * 32 + threadIdx.x;
  for (int r = threadIdx.y; r < 32; r += 8) {
    int y = blockIdx.y * 32 + r;
    if (x < NPIX && y < NPIX) tile[r][threadIdx.x] = src[(size_t)y * NPIX + x];
  }
  __syncthreads();
  int tx = blockIdx.y * 32 + threadIdx.x;
  for (int r = threadIdx.y; r < 32; r += 8) {
    int ty = blockIdx.x * 32 + r;
    if (tx < NPIX && ty < NPIX) dst[(size_t)ty * NPIX + tx] = tile[threadIdx.x][r];
  }
}

// Kf = conj(X) * Y / (|X|^2 + eps), in place into X
__global__ void k_kf(float2* __restrict__ X, const float2* __restrict__ Y) {
  int stride = gridDim.x * blockDim.x;
  for (int i = blockIdx.x * blockDim.x + threadIdx.x; i < NPIX2; i += stride) {
    float2 x = X[i], y = Y[i];
    float den = x.x * x.x + x.y * x.y + EPSF;
    X[i] = make_float2((x.x * y.x + x.y * y.y) / den,
                       (x.x * y.y - x.y * y.x) / den);
  }
}

// Partial inverse DFT over u: T1[a][v] = sum_u Z[v][u] * e^{+2pi i u (CH+a)/N}
// Z is the transposed-layout spectrum: Z[v*N + u] = Kf[u, v].
__global__ __launch_bounds__(256)
void k_t1(const float2* __restrict__ Z, float2* __restrict__ T1) {
  __shared__ float2 red[256];
  const int v = blockIdx.x;
  const float2* zrow = Z + (size_t)v * NPIX;
  float2 acc[KS];
#pragma unroll
  for (int a = 0; a < KS; ++a) acc[a] = make_float2(0.f, 0.f);
  for (int u = threadIdx.x; u < NPIX; u += 256) {
    float2 z = zrow[u];
#pragma unroll
    for (int a = 0; a < KS; ++a) {
      int ph = (int)(((long long)u * (CH + a)) % NPIX);
      float ang = 2.f * PI_F * (float)ph / (float)NPIX;
      float s, c; __sincosf(ang, &s, &c);
      acc[a].x += z.x * c - z.y * s;
      acc[a].y += z.x * s + z.y * c;
    }
  }
  for (int a = 0; a < KS; ++a) {
    red[threadIdx.x] = acc[a];
    __syncthreads();
    for (int off = 128; off > 0; off >>= 1) {
      if (threadIdx.x < off) {
        red[threadIdx.x].x += red[threadIdx.x + off].x;
        red[threadIdx.x].y += red[threadIdx.x + off].y;
      }
      __syncthreads();
    }
    if (threadIdx.x == 0) T1[(size_t)a * NPIX + v] = red[0];
    __syncthreads();
  }
}

// kn[a][b] = (1/N^2) Re sum_v T1[a][v] e^{+2pi i v (CH+b)/N}; relu; normalize
__global__ __launch_bounds__(128)
void k_estimate_ker(const float2* __restrict__ T1,
                    float* __restrict__ ker, float* __restrict__ kerflip) {
  __shared__ float kn[KS * KS];
  __shared__ float ssum;
  int t = threadIdx.x;
  if (t < KS * KS) {
    int a = t / KS, b = t % KS;
    const float2* ta = T1 + (size_t)a * NPIX;
    float acc = 0.f;
    for (int v = 0; v < NPIX; ++v) {
      int ph = (int)(((long long)v * (CH + b)) % NPIX);
      float ang = 2.f * PI_F * (float)ph / (float)NPIX;
      float s, c; __sincosf(ang, &s, &c);
      float2 tv = ta[v];
      acc += tv.x * c - tv.y * s;
    }
    acc /= (float)NPIX * (float)NPIX;
    kn[t] = acc > 0.f ? acc : 0.f;
  }
  __syncthreads();
  if (t == 0) {
    float s = 0.f;
    for (int i = 0; i < KS * KS; ++i) s += kn[i];
    ssum = s + EPSF;
  }
  __syncthreads();
  if (t < KS * KS) {
    float v = kn[t] / ssum;
    ker[t] = v;
    kerflip[KS * KS - 1 - t] = v;
  }
}

__global__ void k_finalize(const float* __restrict__ x32, const float* __restrict__ ker,
                           float* __restrict__ out) {
  int stride = gridDim.x * blockDim.x;
  for (int i = blockIdx.x * blockDim.x + threadIdx.x; i < NPIX2; i += stride)
    out[i] = x32[i];
  if (blockIdx.x == 0 && threadIdx.x == 0) {
    float s = 0.f;
    for (int j = 0; j < KS * KS; ++j) s += ker[j];
    float inv = 1.f / s;
    for (int j = 0; j < KS * KS; ++j) out[NPIX2 + j] = ker[j] * inv;
  }
}

// ---------------------------------------------------------------------------
extern "C" void kernel_launch(void* const* d_in, const int* in_sizes, int n_in,
                              void* d_out, int out_size, void* d_ws, size_t ws_size,
                              hipStream_t stream) {
  (void)in_sizes; (void)n_in; (void)out_size; (void)ws_size;
  const float* blurred = (const float*)d_in[0];

  char* p = (char*)d_ws;
  auto alloc = [&](size_t bytes) { void* r = (void*)p; p += (bytes + 255) & ~(size_t)255; return r; };
  float*  x32    = (float*) alloc((size_t)NPIX2 * sizeof(float));
  float2* Ybuf   = (float2*)alloc((size_t)NPIX2 * sizeof(float2));
  float2* Xbuf   = (float2*)alloc((size_t)NPIX2 * sizeof(float2));
  float2* Cs     = (float2*)alloc((size_t)NPIX2 * sizeof(float2)); // FFT scratch
  float2* chirpW = (float2*)alloc((size_t)NPIX * sizeof(float2));
  float2* bbuf   = (float2*)alloc((size_t)MFFT * sizeof(float2));
  float2* Bhat   = (float2*)alloc((size_t)MFFT * sizeof(float2));
  float2* T1     = (float2*)alloc((size_t)KS * NPIX * sizeof(float2));
  float*  ker    = (float*) alloc(KS * KS * sizeof(float));
  float*  kerflip= (float*) alloc(KS * KS * sizeof(float));
  // RL temporaries overlay the FFT scratch buffer (disjoint phases):
  float*          tmp32 = (float*)Cs;                                   // conv output (f32)
  unsigned short* xbf   = (unsigned short*)((char*)Cs + (size_t)NPIX2 * sizeof(float)); // bf16, pitch XBF_W

  const dim3 convGrid(24, 192), convBlock(256);     // 24*128 x 192*16 tiles cover 3071^2
  const dim3 trGrid(96, 96), trBlock(32, 8);
  const int  psBlocks = 2048;

  k_init_ker<<<1, 128, 0, stream>>>(ker, kerflip);
  k_chirp_setup<<<(MFFT + 255) / 256, 256, 0, stream>>>(chirpW, bbuf);
  k_fft8192<<<1, 256, 0, stream>>>(bbuf, Bhat, -1);

  // Y = fft2(blurred), stored transposed: Y[v][u]
  k_bluestein_rows<<<NPIX, 256, 0, stream>>>((const void*)blurred, 1, Cs, chirpW, Bhat);
  k_transpose<<<trGrid, trBlock, 0, stream>>>(Cs, Ybuf);
  k_bluestein_rows<<<NPIX, 256, 0, stream>>>((const void*)Ybuf, 0, Ybuf, chirpW, Bhat);

  for (int it = 0; it < NUM_ITERS; ++it) {
    k_init_from_blurred<<<psBlocks, 256, 0, stream>>>(blurred, x32, xbf);
    for (int rl = 0; rl < RL_ITERS; ++rl) {
      k_conv_wmma<<<convGrid, convBlock, 0, stream>>>(xbf, tmp32, ker);      // est
      k_ratio<<<psBlocks, 256, 0, stream>>>(blurred, tmp32, xbf);            // ratio (bf16)
      k_conv_wmma<<<convGrid, convBlock, 0, stream>>>(xbf, tmp32, kerflip);  // corr
      k_update_x<<<psBlocks, 256, 0, stream>>>(x32, tmp32, xbf);             // x = clip(x*corr)
    }
    // X = fft2(x) (transposed layout), then Kf, then direct center-crop inverse DFT
    k_bluestein_rows<<<NPIX, 256, 0, stream>>>((const void*)x32, 1, Cs, chirpW, Bhat);
    k_transpose<<<trGrid, trBlock, 0, stream>>>(Cs, Xbuf);
    k_bluestein_rows<<<NPIX, 256, 0, stream>>>((const void*)Xbuf, 0, Xbuf, chirpW, Bhat);
    k_kf<<<psBlocks, 256, 0, stream>>>(Xbuf, Ybuf);
    k_t1<<<NPIX, 256, 0, stream>>>(Xbuf, T1);
    k_estimate_ker<<<1, 128, 0, stream>>>(T1, ker, kerflip);
  }
  k_finalize<<<psBlocks, 256, 0, stream>>>(x32, ker, (float*)d_out);
}
